// ECGTokenizer_53420803228140
// MI455X (gfx1250) — matrix operand order repaired and checked by reading, not compiled, not run on hardware
//
#include <hip/hip_runtime.h>

// ---------------------------------------------------------------------------
// ECG beat tokenizer for MI455X (gfx1250): GEMM M=196608, K=128, N=64.
// fp32 -> bf16 hi/lo split -> 3x v_wmma_f32_16x16x32_bf16 (fp32-class acc).
// W is pre-converted ONCE into WMMA B-fragment layout (prep kernel -> d_ws),
// TDM stages both the ecg tile (with LDS row padding for conflict-free
// ds_load_b128) and the B-fragment blob into LDS.
// ---------------------------------------------------------------------------

typedef __attribute__((ext_vector_type(16))) __bf16        bf16x16;
typedef __attribute__((ext_vector_type(8)))  float         floatx8;
typedef __attribute__((ext_vector_type(4)))  unsigned int  uint32x4;
typedef __attribute__((ext_vector_type(8)))  int           int32x8;
typedef __attribute__((ext_vector_type(4)))  int           int32x4;

#define BEAT_LEN    128
#define TOKEN_DIM   64
#define ROWS_PER_WG 64
#define LDS_STRIDE  132       // 128 + 4-dword TDM pad -> 4-bank lane stride
#define BFRAG_DW    8192      // 16 combos * 2 (hi/lo) * 32 lanes * 8 dwords

union Frag {
  unsigned u[8];
  bf16x16  v;
};

// --- cheap bf16 hi/lo split ------------------------------------------------
// hi: round-to-nearest (+0x8000) top 16 bits; lo: truncated residual.
// Packing via v_perm_b32: result = {hi16(b), hi16(a)}.
__device__ __forceinline__ unsigned pack_hi16(unsigned b, unsigned a) {
  return __builtin_amdgcn_perm(b, a, 0x07060302u);
}
__device__ __forceinline__ void split2(float a, float b, unsigned& hi, unsigned& lo) {
  unsigned ua = __float_as_uint(a), ub = __float_as_uint(b);
  unsigned ra = ua + 0x8000u, rb = ub + 0x8000u;           // round-nearest
  hi = pack_hi16(rb, ra);
  float alo = a - __uint_as_float(ra & 0xFFFF0000u);
  float blo = b - __uint_as_float(rb & 0xFFFF0000u);
  lo = pack_hi16(__float_as_uint(blo), __float_as_uint(alo));
}
// 8 consecutive floats -> 4 packed hi dwords + 4 packed lo dwords
__device__ __forceinline__ void cvt_run8(const float* p, unsigned* uh, unsigned* ul) {
  float4 x = *(const float4*)(p);
  float4 y = *(const float4*)(p + 4);
  split2(x.x, x.y, uh[0], ul[0]);
  split2(x.z, x.w, uh[1], ul[1]);
  split2(y.x, y.y, uh[2], ul[2]);
  split2(y.z, y.w, uh[3], ul[3]);
}

// --- TDM tile load (data_size = 4B), optional 4-dword pad per 128 dwords ---
__device__ __forceinline__ void tdm_load(unsigned lds_addr, const void* gptr,
                                         unsigned dim0, unsigned dim1,
                                         unsigned stride0, bool pad) {
  unsigned long long ga = (unsigned long long)(size_t)gptr;
  uint32x4 g0 = {
      1u,                                                   // count=1, user desc
      lds_addr,                                             // lds_addr[31:0]
      (unsigned)(ga & 0xFFFFFFFFu),                         // global_addr[31:0]
      (unsigned)((ga >> 32) & 0x01FFFFFFu) | (2u << 30)     // ga[56:32] | type=2
  };
  unsigned flags = 2u << 16;                                // data_size=4B
  if (pad) flags |= (1u << 20) | (6u << 22) | (3u << 25);   // 4 dw pad / 128 dw
  int32x8 g1 = {
      (int)flags,
      (int)((dim0 & 0xFFFFu) << 16),                                // tensor_dim0 lo
      (int)(((dim0 >> 16) & 0xFFFFu) | ((dim1 & 0xFFFFu) << 16)),   // tdim0 hi|tdim1 lo
      (int)(((dim1 >> 16) & 0xFFFFu) | ((dim0 & 0xFFFFu) << 16)),   // tdim1 hi|tile_dim0
      (int)(dim1 & 0xFFFFu),                                        // tile_dim1, tile_dim2=0
      (int)stride0,                                                 // tensor_dim0_stride
      0, 0
  };
  int32x4 gz4 = {0, 0, 0, 0};
  int32x8 gz8 = {0, 0, 0, 0, 0, 0, 0, 0};
  __builtin_amdgcn_tensor_load_to_lds(g0, g1, gz4, gz4, gz8, 0);
}

// ---------------------------------------------------------------------------
// Prep kernel (32 threads/block):
//   blocks 0..15  : convert W into B-fragment layout in d_ws
//                   layout dword index: ((combo*2 + h)*32 + lane)*8
//   blocks 16..   : fill beat_intervals with 128.0f (32 elems/block)
// ---------------------------------------------------------------------------
__global__ __launch_bounds__(32) void prep_kernel(
    const float* __restrict__ W, unsigned* __restrict__ bfrag,
    float* __restrict__ beat, int beatN) {
  const int lane = threadIdx.x;
  if (blockIdx.x < 16) {
    const int combo = blockIdx.x;             // j*4 + c
    const int j = combo >> 2, c = combo & 3;
    const int half = lane >> 4, l16 = lane & 15;
    // B[k][n] = W[n][k]; lane = column n; lanes 0-15: K=c*32+{0..7,16..23}, 16-31: +8
    const float* wp = W + (j * 16 + l16) * BEAT_LEN + c * 32 + half * 8;
    Frag hi, lo;
    cvt_run8(wp,     hi.u,     lo.u);
    cvt_run8(wp + 16, hi.u + 4, lo.u + 4);
    unsigned dhi = (unsigned)(((combo * 2 + 0) * 32 + lane) * 8);
    unsigned dlo = (unsigned)(((combo * 2 + 1) * 32 + lane) * 8);
    *(uint4*)(bfrag + dhi)     = *(const uint4*)&hi.u[0];
    *(uint4*)(bfrag + dhi + 4) = *(const uint4*)&hi.u[4];
    *(uint4*)(bfrag + dlo)     = *(const uint4*)&lo.u[0];
    *(uint4*)(bfrag + dlo + 4) = *(const uint4*)&lo.u[4];
  } else {
    const int i = (blockIdx.x - 16) * 32 + lane;
    if (i < beatN) beat[i] = 128.0f;          // float(BEAT_LEN)
  }
}

// ---------------------------------------------------------------------------
// Main GEMM kernel: 128 threads = 4 waves, 64 rows x 64 cols per block.
// ---------------------------------------------------------------------------
__global__ __launch_bounds__(128) void ecg_tok_kernel(
    const float* __restrict__ ecg, const unsigned* __restrict__ bfrag,
    const float* __restrict__ bias, float* __restrict__ X) {
  __shared__ __align__(16) float    ldsEcg[ROWS_PER_WG * LDS_STRIDE];  // 33792 B
  __shared__ __align__(16) unsigned ldsB[BFRAG_DW];                    // 32768 B

  const unsigned rowBase = blockIdx.x * ROWS_PER_WG;
  const unsigned wave    = threadIdx.x >> 5;

  if (wave == 0) {  // wave-uniform branch; TDM ignores EXEC
    tdm_load((unsigned)(size_t)(void*)ldsEcg,
             ecg + (size_t)rowBase * BEAT_LEN,
             BEAT_LEN, ROWS_PER_WG, BEAT_LEN, /*pad=*/true);
    tdm_load((unsigned)(size_t)(void*)ldsB,
             bfrag, BFRAG_DW, 1, BFRAG_DW, /*pad=*/false);
    __builtin_amdgcn_s_wait_tensorcnt(0);
  }
  __syncthreads();

  const unsigned lane = threadIdx.x & 31;
  const unsigned half = lane >> 4;
  const unsigned l16  = lane & 15;

  // ---- A fragments (16 rows x 128 K), bf16 hi/lo, per 16-bit A layout:
  // lanes 0-15 : row=l16, K = c*32 + {0..7} in v0..3, c*32 + {16..23} in v4..7
  // lanes 16-31: row=l16, K = c*32 + {8..15},  c*32 + {24..31}
  Frag aHi[4], aLo[4];
  const float* arow = &ldsEcg[(wave * 16 + l16) * LDS_STRIDE];
#pragma unroll
  for (int c = 0; c < 4; ++c) {
    const unsigned k0 = c * 32 + half * 8;
    cvt_run8(arow + k0,      aHi[c].u,     aLo[c].u);
    cvt_run8(arow + k0 + 16, aHi[c].u + 4, aLo[c].u + 4);
  }

  // ---- 4 column tiles of 16; pre-swizzled B fragments straight from LDS.
#pragma unroll
  for (int j = 0; j < 4; ++j) {
    const unsigned col = j * 16 + l16;
    floatx8 acc = {0.f, 0.f, 0.f, 0.f, 0.f, 0.f, 0.f, 0.f};
#pragma unroll
    for (int c = 0; c < 4; ++c) {
      const unsigned* bp = &ldsB[((j * 4 + c) * 2 * 32 + lane) * 8];
      Frag bHi, bLo;
      *(uint4*)&bHi.u[0] = *(const uint4*)(bp);
      *(uint4*)&bHi.u[4] = *(const uint4*)(bp + 4);
      *(uint4*)&bLo.u[0] = *(const uint4*)(bp + 256);   // +32 lanes * 8 dw
      *(uint4*)&bLo.u[4] = *(const uint4*)(bp + 260);
      // bf16x3: hi*hi + hi*lo + lo*hi (f32 accumulate)
      acc = __builtin_amdgcn_wmma_f32_16x16x32_bf16(false, aHi[c].v, false, bHi.v,
                                                    (short)0, acc, false, false);
      acc = __builtin_amdgcn_wmma_f32_16x16x32_bf16(false, aHi[c].v, false, bLo.v,
                                                    (short)0, acc, false, false);
      acc = __builtin_amdgcn_wmma_f32_16x16x32_bf16(false, aLo[c].v, false, bHi.v,
                                                    (short)0, acc, false, false);
    }
    // C/D layout: VGPR r -> M = r + (half?8:0), N = l16
    const float    bv   = bias[col];
    const unsigned mRow = rowBase + wave * 16 + half * 8;
#pragma unroll
    for (int r = 0; r < 8; ++r) {
      X[(size_t)(mRow + r) * TOKEN_DIM + col] = acc[r] + bv;
    }
  }
}

extern "C" void kernel_launch(void* const* d_in, const int* in_sizes, int n_in,
                              void* d_out, int out_size, void* d_ws, size_t ws_size,
                              hipStream_t stream) {
  const float* ecg  = (const float*)d_in[0];   // [B, L, T] f32
  const float* W    = (const float*)d_in[1];   // [64, 128] f32
  const float* bias = (const float*)d_in[2];   // [64] f32
  float* out = (float*)d_out;                  // X flat ++ beat_intervals flat
  unsigned* bfrag = (unsigned*)d_ws;           // 32 KB B-fragment blob

  const int M         = in_sizes[0] / BEAT_LEN;  // 196608 rows
  const int xElems    = M * TOKEN_DIM;           // 12,582,912
  const int beatElems = out_size - xElems;       // 16,384

  // prep: blocks 0-15 convert W -> fragments; rest fill beat_intervals
  const int fillBlocks = (beatElems + 31) / 32;
  prep_kernel<<<16 + fillBlocks, 32, 0, stream>>>(W, bfrag, out + xElems,
                                                  beatElems);

  const int blocks = M / ROWS_PER_WG;            // 3072 workgroups
  ecg_tok_kernel<<<blocks, 128, 0, stream>>>(ecg, bfrag, bias, out);
}
